// MultiHeadAttention_5626407158462
// MI455X (gfx1250) — matrix operand order repaired and checked by reading, not compiled, not run on hardware
//
#include <hip/hip_runtime.h>

#define Bn 16
#define Sn 1024
#define Dn 384
#define Hn 6
#define DHn 64
#define Nn (Bn * Sn)        // 16384 rows
#define C3 (3 * Hn * DHn)   // 1152 packed qkv columns

typedef __attribute__((ext_vector_type(16))) _Float16 v16h;
typedef __attribute__((ext_vector_type(8)))  _Float16 v8h;
typedef __attribute__((ext_vector_type(8)))  float    v8f;

#define WMMA_F16(a, b, c) \
  __builtin_amdgcn_wmma_f32_16x16x32_f16(false, (a), false, (b), (short)0, (c), false, false)

// A-operand loader for v_wmma_f32_16x16x32_f16.
// Lane L in [0,16):  row = L,    k0..k0+15
// Lane L in [16,32): row = L-16, k0+16..k0+31
// p = row base + k0 ; hi = lane>>4
__device__ __forceinline__ v16h load_a(const _Float16* p, int hi) {
  union { v16h v; v8h h[2]; } u;
  u.h[0] = *(const v8h*)(p + hi * 16);
  u.h[1] = *(const v8h*)(p + hi * 16 + 8);
  return u.v;
}

// ---------------------------------------------------------------------------
// Kernel 0: precision conversion + weight packing
//   xh   [N][384]    f16 <- x
//   wall [1152][384] f16 (packed W, row c = output col: contiguous B-operand)
//   ball [1152]      f32
//   woh  [384][384]  f16 <- Wo ([out][in] row-major = B-operand layout)
// ---------------------------------------------------------------------------
__global__ void mha_pack(const float* __restrict__ x,
                         const float* __restrict__ Wq, const float* __restrict__ bq,
                         const float* __restrict__ Wk, const float* __restrict__ bk,
                         const float* __restrict__ Wv, const float* __restrict__ bv,
                         const float* __restrict__ Wo,
                         _Float16* __restrict__ xh, _Float16* __restrict__ wall,
                         float* __restrict__ ball, _Float16* __restrict__ woh) {
  size_t tid = (size_t)blockIdx.x * 256 + threadIdx.x;
  if (tid < (size_t)Nn * Dn) xh[tid] = (_Float16)x[tid];
  if (tid < (size_t)C3 * Dn) {
    int c = (int)(tid / Dn), d = (int)(tid % Dn);
    int sel = c / (Hn * DHn);
    int cl = c % (Hn * DHn);
    int h = cl / DHn, e = cl % DHn;
    const float* W = (sel == 0) ? Wq : (sel == 1) ? Wk : Wv;
    wall[(size_t)c * Dn + d] = (_Float16)W[((size_t)h * Dn + d) * DHn + e];
  }
  if (tid < C3) {
    int sel = (int)tid / (Hn * DHn);
    int cl = (int)tid % (Hn * DHn);
    int h = cl / DHn, e = cl % DHn;
    const float* bp = (sel == 0) ? bq : (sel == 1) ? bk : bv;
    ball[tid] = bp[h * DHn + e];
  }
  if (tid < (size_t)Dn * Dn) woh[tid] = (_Float16)Wo[tid];
}

// ---------------------------------------------------------------------------
// Kernel 1: QKV projection GEMM  [16384 x 1152] = xh * wall^T (+bias)
//   Block = 64 rows x 128 cols (4 waves). A panel (64x384, contiguous 48 KB)
//   staged in LDS once; wave owns 2 col tiles x 4 row tiles (8 accумs).
//   Per K-step: 2 global B loads + 4 LDS A frags -> 8 WMMAs.
//   Q,K stored [B][H][S][64]; V stored transposed [B][H][64][S].
// ---------------------------------------------------------------------------
__global__ __launch_bounds__(128) void mha_qkv(
    const _Float16* __restrict__ xh, const _Float16* __restrict__ wall,
    const float* __restrict__ ball,
    _Float16* __restrict__ qh, _Float16* __restrict__ kh, _Float16* __restrict__ vt) {
  __shared__ __align__(64) _Float16 sA[64 * Dn];   // 48 KB

  const int bm = blockIdx.x / (C3 / 128);          // 0..255 row block
  const int bc = blockIdx.x % (C3 / 128);          // 0..8   col block
  const int R0 = bm * 64;
  const int lane = threadIdx.x & 31, wave = threadIdx.x >> 5;
  const int r15 = lane & 15, hi = lane >> 4;

  // stage contiguous 64x384 A panel into LDS
  const _Float16* gpanel = xh + (size_t)R0 * Dn;
  for (int i = threadIdx.x * 8; i < 64 * Dn; i += 128 * 8)
    *(v8h*)(sA + i) = *(const v8h*)(gpanel + i);
  __syncthreads();

  const int cw = bc * 128 + wave * 32;             // wave's first column
  v8f acc[4][2] = {{{}, {}}, {{}, {}}, {{}, {}}, {{}, {}}};
  const _Float16* brow0 = wall + (size_t)(cw + r15) * Dn + hi * 16;
  const _Float16* brow1 = wall + (size_t)(cw + 16 + r15) * Dn + hi * 16;

  for (int k0 = 0; k0 < Dn; k0 += 32) {
    v16h b0 = *(const v16h*)(brow0 + k0);
    v16h b1 = *(const v16h*)(brow1 + k0);
#pragma unroll
    for (int m = 0; m < 4; ++m) {
      v16h a = load_a(sA + (m * 16 + r15) * Dn + k0, hi);
      acc[m][0] = WMMA_F16(a, b0, acc[m][0]);
      acc[m][1] = WMMA_F16(a, b1, acc[m][1]);
    }
  }

#pragma unroll
  for (int cb = 0; cb < 2; ++cb) {
    const int cg = cw + cb * 16 + r15;             // packed column
    const float bias = ball[cg];
    const int sel = cg / (Hn * DHn);               // uniform per wave (32|384)
    const int cl = cg % (Hn * DHn);
    const int h = cl >> 6, e = cl & 63;
#pragma unroll
    for (int m = 0; m < 4; ++m) {
      const int rbase = R0 + m * 16 + (hi ? 8 : 0);
      for (int j = 0; j < 8; ++j) {
        int n = rbase + j;
        int b = n >> 10, s = n & (Sn - 1);
        _Float16 hv = (_Float16)(acc[m][cb][j] + bias);
        size_t bh = (size_t)(b * Hn + h);
        if (sel == 0)      qh[(bh * Sn + s) * DHn + e] = hv;
        else if (sel == 1) kh[(bh * Sn + s) * DHn + e] = hv;
        else               vt[(bh * DHn + e) * Sn + s] = hv;
      }
    }
  }
}

// ---------------------------------------------------------------------------
// Kernel 2: attention. One workgroup (4 waves) per (b,h, 16-row query block).
//   Pass1: scores (QK^T/8) -> LDS f32 [16][1024]; Q frags hoisted, K streamed.
//   Softmax cooperatively (8 threads per row), P -> LDS f16 [16][1024].
//   Pass2: ctx = P * V, each wave owns 16 of the 64 head cols, 32 WMMAs.
// ---------------------------------------------------------------------------
__global__ __launch_bounds__(128) void mha_attn(
    const _Float16* __restrict__ qh, const _Float16* __restrict__ kh,
    const _Float16* __restrict__ vt, _Float16* __restrict__ cc) {
  __shared__ __align__(64) float    sS[16 * Sn];   // 64 KB scores
  __shared__ __align__(64) _Float16 sP[16 * Sn];   // 32 KB probs
  __shared__ float red[16 * 8];
  __shared__ float rowstat[16];

  const int bh = blockIdx.x >> 6;                  // b*H + h
  const int q0 = (blockIdx.x & 63) * 16;
  const int lane = threadIdx.x & 31;
  const int wave = threadIdx.x >> 5;
  const int r15 = lane & 15, hi = lane >> 4;
  const size_t base = (size_t)bh * Sn * DHn;

  // ---- pass 1: scores (Q fragments are loop-invariant: hoist) ----
  const _Float16* qrow = qh + base + (size_t)(q0 + r15) * DHn;
  const v16h qa0 = load_a(qrow + 0, hi);
  const v16h qa1 = load_a(qrow + 32, hi);
  for (int tt = wave; tt < Sn / 16; tt += 4) {
    int t0 = tt * 16;
    const _Float16* krow = kh + base + (size_t)(t0 + r15) * DHn + hi * 16;
    v16h kb0 = *(const v16h*)(krow + 0);
    v16h kb1 = *(const v16h*)(krow + 32);
    v8f acc = {};
    acc = WMMA_F16(qa0, kb0, acc);
    acc = WMMA_F16(qa1, kb1, acc);
    int rb = hi ? 8 : 0;
    for (int j = 0; j < 8; ++j)
      sS[(rb + j) * Sn + t0 + r15] = acc[j] * 0.125f;  // 1/sqrt(64)
  }
  __syncthreads();

  // ---- softmax: 8 threads per row ----
  const int r = threadIdx.x >> 3, sub = threadIdx.x & 7;
  float* srow = sS + r * Sn;
  float mx = -3.4e38f;
  for (int c0 = sub * 128; c0 < sub * 128 + 128; ++c0) mx = fmaxf(mx, srow[c0]);
  red[r * 8 + sub] = mx;
  __syncthreads();
  if (sub == 0) {
    float m2 = red[r * 8];
    for (int t = 1; t < 8; ++t) m2 = fmaxf(m2, red[r * 8 + t]);
    rowstat[r] = m2;
  }
  __syncthreads();
  const float rm = rowstat[r];
  float sum = 0.f;
  for (int c0 = sub * 128; c0 < sub * 128 + 128; ++c0) {
    float ev = __expf(srow[c0] - rm);
    srow[c0] = ev;
    sum += ev;
  }
  red[r * 8 + sub] = sum;
  __syncthreads();
  if (sub == 0) {
    float s2 = 0.f;
    for (int t = 0; t < 8; ++t) s2 += red[r * 8 + t];
    rowstat[r] = 1.0f / s2;
  }
  __syncthreads();
  const float inv = rowstat[r];
  for (int c0 = sub * 128; c0 < sub * 128 + 128; ++c0)
    sP[r * Sn + c0] = (_Float16)(srow[c0] * inv);
  __syncthreads();

  // ---- pass 2: ctx = P * V (wave owns cols e0..e0+15) ----
  {
    const int e0 = wave * 16;
    v8f acc = {};
    const _Float16* prow = sP + r15 * Sn;
    const _Float16* vrow = vt + (size_t)bh * DHn * Sn + (size_t)(e0 + r15) * Sn + hi * 16;
#pragma unroll 4
    for (int kt = 0; kt < Sn; kt += 32) {
      v16h a = load_a(prow + kt, hi);
      v16h b = *(const v16h*)(vrow + kt);
      acc = WMMA_F16(a, b, acc);
    }
    const int bb = bh / Hn, h = bh % Hn;
    const int rb = hi ? 8 : 0;
    for (int j = 0; j < 8; ++j) {
      size_t rg = (size_t)bb * Sn + q0 + rb + j;
      cc[rg * Dn + h * DHn + e0 + r15] = (_Float16)acc[j];
    }
  }
}

// ---------------------------------------------------------------------------
// Kernel 3: output projection  out[16384x384] = cc * Wo^T + bo   (f32 out)
//   Same 64x128 block structure as the QKV GEMM.
// ---------------------------------------------------------------------------
__global__ __launch_bounds__(128) void mha_proj(
    const _Float16* __restrict__ cc, const _Float16* __restrict__ woh,
    const float* __restrict__ bo, float* __restrict__ out) {
  __shared__ __align__(64) _Float16 sA[64 * Dn];   // 48 KB

  const int bm = blockIdx.x / (Dn / 128);          // 0..255
  const int bc = blockIdx.x % (Dn / 128);          // 0..2
  const int R0 = bm * 64;
  const int lane = threadIdx.x & 31, wave = threadIdx.x >> 5;
  const int r15 = lane & 15, hi = lane >> 4;

  const _Float16* gpanel = cc + (size_t)R0 * Dn;
  for (int i = threadIdx.x * 8; i < 64 * Dn; i += 128 * 8)
    *(v8h*)(sA + i) = *(const v8h*)(gpanel + i);
  __syncthreads();

  const int cw = bc * 128 + wave * 32;
  v8f acc[4][2] = {{{}, {}}, {{}, {}}, {{}, {}}, {{}, {}}};
  const _Float16* brow0 = woh + (size_t)(cw + r15) * Dn + hi * 16;
  const _Float16* brow1 = woh + (size_t)(cw + 16 + r15) * Dn + hi * 16;

  for (int k0 = 0; k0 < Dn; k0 += 32) {
    v16h b0 = *(const v16h*)(brow0 + k0);
    v16h b1 = *(const v16h*)(brow1 + k0);
#pragma unroll
    for (int m = 0; m < 4; ++m) {
      v16h a = load_a(sA + (m * 16 + r15) * Dn + k0, hi);
      acc[m][0] = WMMA_F16(a, b0, acc[m][0]);
      acc[m][1] = WMMA_F16(a, b1, acc[m][1]);
    }
  }

#pragma unroll
  for (int cb = 0; cb < 2; ++cb) {
    const int col = cw + cb * 16 + r15;
    const float bias = bo[col];
#pragma unroll
    for (int m = 0; m < 4; ++m) {
      const int rbase = R0 + m * 16 + (hi ? 8 : 0);
      for (int j = 0; j < 8; ++j)
        out[(size_t)(rbase + j) * Dn + col] = acc[m][cb][j] + bias;
    }
  }
}

// ---------------------------------------------------------------------------
extern "C" void kernel_launch(void* const* d_in, const int* in_sizes, int n_in,
                              void* d_out, int out_size, void* d_ws, size_t ws_size,
                              hipStream_t stream) {
  const float* x  = (const float*)d_in[0];
  const float* Wq = (const float*)d_in[1];
  const float* bq = (const float*)d_in[2];
  const float* Wk = (const float*)d_in[3];
  const float* bk = (const float*)d_in[4];
  const float* Wv = (const float*)d_in[5];
  const float* bv = (const float*)d_in[6];
  const float* Wo = (const float*)d_in[7];
  const float* bo = (const float*)d_in[8];
  float* out = (float*)d_out;

  char* ws = (char*)d_ws;
  size_t off = 0;
  auto nxt = [&](size_t bytes) {
    size_t o = off;
    off = (off + bytes + 255) & ~(size_t)255;
    return o;
  };
  _Float16* xh   = (_Float16*)(ws + nxt((size_t)Nn * Dn * 2));
  _Float16* wall = (_Float16*)(ws + nxt((size_t)C3 * Dn * 2));
  float*    ball = (float*)   (ws + nxt((size_t)C3 * 4));
  _Float16* qh   = (_Float16*)(ws + nxt((size_t)Bn * Hn * Sn * DHn * 2));
  _Float16* kh   = (_Float16*)(ws + nxt((size_t)Bn * Hn * Sn * DHn * 2));
  _Float16* vt   = (_Float16*)(ws + nxt((size_t)Bn * Hn * Sn * DHn * 2));
  _Float16* ccb  = (_Float16*)(ws + nxt((size_t)Nn * Dn * 2));
  _Float16* woh  = (_Float16*)(ws + nxt((size_t)Dn * Dn * 2));

  // pack / convert
  mha_pack<<<(Nn * Dn) / 256, 256, 0, stream>>>(x, Wq, bq, Wk, bk, Wv, bv, Wo,
                                                xh, wall, ball, woh);
  // qkv: (16384/64) row blocks * (1152/128) col blocks = 2304 blocks
  mha_qkv<<<(Nn / 64) * (C3 / 128), 128, 0, stream>>>(xh, wall, ball, qh, kh, vt);
  // attention: B*H*(S/16) = 6144 blocks
  mha_attn<<<Bn * Hn * (Sn / 16), 128, 0, stream>>>(qh, kh, vt, ccb);
  // projection: (16384/64) * (384/128) = 768 blocks
  mha_proj<<<(Nn / 64) * (Dn / 128), 128, 0, stream>>>(ccb, woh, bo, out);
}
